// DynamicFSFilter_76029511073935
// MI455X (gfx1250) — matrix operand order repaired and checked by reading, not compiled, not run on hardware
//
#include <hip/hip_runtime.h>
#include <math.h>

// ---------------------------------------------------------------------------
// CDNA5 / gfx1250 implementation of the DynamicFSFilter block.
// wave32, WMMA 16x16x32 f16 for all GEMMs; FFT planes resident in LDS.
// BK=64: 4 v_wmma per wave per barrier interval; vectorized b128 staging.
// ---------------------------------------------------------------------------

typedef _Float16 f16;
typedef __attribute__((ext_vector_type(16))) _Float16 v16h;
typedef __attribute__((ext_vector_type(8)))  _Float16 v8h;
typedef __attribute__((ext_vector_type(4)))  _Float16 h4;
typedef __attribute__((ext_vector_type(8)))  float    v8f;

#define DIMC 512
#define MEDC 1024
#define NPIX 4096          // 64*64
#define FSZ  33            // rfft width
#define HRC  256
#define HSC  256
#define NB   16

#define BM 64
#define BN 64
#define BK 64
#define KS 72              // padded k-stride (halves): 64 + 8

__device__ __forceinline__ v8f wmma_f16(v16h a, v16h b, v8f c) {
  // D = A(16x32) * B(32x16) + C  -> v_wmma_f32_16x16x32_f16
  return __builtin_amdgcn_wmma_f32_16x16x32_f16(
      false, a, false, b, (short)0, c, false, false);
}

// A fragment from [m][k] LDS: one contiguous 16-half run (k = 16*half + s).
__device__ __forceinline__ v16h load_a_frag(const f16* rowBase, int half) {
  v8h lo = *(const v8h*)(rowBase + half * 16);
  v8h hi = *(const v8h*)(rowBase + half * 16 + 8);
  return __builtin_shufflevector(lo, hi, 0,1,2,3,4,5,6,7,8,9,10,11,12,13,14,15);
}
// B fragment from transposed [n][k] LDS: one contiguous 16-half run.
__device__ __forceinline__ v16h load_b_frag(const f16* rowBase, int half) {
  v8h lo = *(const v8h*)(rowBase + half * 16);
  v8h hi = *(const v8h*)(rowBase + half * 16 + 8);
  return __builtin_shufflevector(lo, hi, 0,1,2,3,4,5,6,7,8,9,10,11,12,13,14,15);
}

constexpr int EPI_BIAS_RELU = 0;
constexpr int EPI_STARRELU  = 1;
constexpr int EPI_NONE      = 2;

// ---------------- generic tiled GEMM: C[M,N] = epi(A[M,K] @ B[K,N]) ---------
// 256 thr = 8 waves; BM=64 x BN=64 x BK=64; each wave: 2 A-frags, 2 N-tiles,
// 4 WMMAs per barrier interval.
template <int EPI>
__global__ __launch_bounds__(256)
void gemm_tiled(const float* __restrict__ A,  long aBatch, int lda,
                const float* __restrict__ Bm, long bBatch, int ldb,
                const float* __restrict__ bias,
                const float* __restrict__ sscale,
                const float* __restrict__ sbias,
                float* __restrict__ C, long cBatch, int ldc,
                int M, int N, int K)
{
  __shared__ __align__(16) f16 As[BM * KS];   // [m][k]
  __shared__ __align__(16) f16 Bs[BN * KS];   // [n][k] transposed

  const int bz = blockIdx.z;
  const float* Ab = A  + (long)bz * aBatch;
  const float* Bb = Bm + (long)bz * bBatch;
  float*       Cb = C  + (long)bz * cBatch;

  const int row0 = blockIdx.y * BM;
  const int col0 = blockIdx.x * BN;
  const int tid  = threadIdx.x;
  const int lane = tid & 31;
  const int wv   = tid >> 5;
  const int tm   = wv >> 1;          // 0..3
  const int tn2  = wv & 1;           // 0..1  (owns n = tn2*32 .. +31)
  const int half = lane >> 4;
  const int mloc = lane & 15;

  v8f acc0 = {0.f,0.f,0.f,0.f,0.f,0.f,0.f,0.f};
  v8f acc1 = {0.f,0.f,0.f,0.f,0.f,0.f,0.f,0.f};

  const f16* rowA  = &As[(tm * 16 + mloc) * KS];
  const f16* rowB0 = &Bs[(tn2 * 32 + mloc) * KS];
  const f16* rowB1 = &Bs[(tn2 * 32 + 16 + mloc) * KS];

  for (int k0 = 0; k0 < K; k0 += BK) {
    if (k0 + BK < K)    // global_prefetch_b8 on the streaming operand
      __builtin_prefetch((const void*)(Bb + (long)(k0 + BK) * ldb + col0 + (tid & 63)), 0, 1);

    // A tile 64x64: 1024 float4, 4 per thread, packed ds_store_b64
#pragma unroll
    for (int j = 0; j < 4; ++j) {
      int idx = tid + j * 256;
      int r  = idx >> 4;
      int c4 = (idx & 15) * 4;
      const float4 v = *(const float4*)&Ab[(long)(row0 + r) * lda + k0 + c4];
      h4 hv; hv[0] = (f16)v.x; hv[1] = (f16)v.y; hv[2] = (f16)v.z; hv[3] = (f16)v.w;
      *(h4*)&As[r * KS + c4] = hv;
    }
    // B tile 64x64 -> transposed [n][k]: 1024 float4, 4 per thread
#pragma unroll
    for (int j = 0; j < 4; ++j) {
      int idx = tid + j * 256;
      int kr = idx >> 4;
      int c4 = (idx & 15) * 4;
      const float4 v = *(const float4*)&Bb[(long)(k0 + kr) * ldb + col0 + c4];
      Bs[(c4 + 0) * KS + kr] = (f16)v.x;
      Bs[(c4 + 1) * KS + kr] = (f16)v.y;
      Bs[(c4 + 2) * KS + kr] = (f16)v.z;
      Bs[(c4 + 3) * KS + kr] = (f16)v.w;
    }
    __syncthreads();

    v16h af0 = load_a_frag(rowA,      half);   // k-chunk 0
    v16h af1 = load_a_frag(rowA + 32, half);   // k-chunk 1
    v16h b00 = load_b_frag(rowB0,      half);
    v16h b01 = load_b_frag(rowB0 + 32, half);
    v16h b10 = load_b_frag(rowB1,      half);
    v16h b11 = load_b_frag(rowB1 + 32, half);
    acc0 = wmma_f16(af0, b00, acc0);
    acc0 = wmma_f16(af1, b01, acc0);
    acc1 = wmma_f16(af0, b10, acc1);
    acc1 = wmma_f16(af1, b11, acc1);
    __syncthreads();
  }

  float ss = 0.f, sb = 0.f;
  if (EPI == EPI_STARRELU) { ss = sscale[0]; sb = sbias[0]; }

#pragma unroll
  for (int v = 0; v < 8; ++v) {
    int m  = row0 + tm * 16 + half * 8 + v;
    int n  = col0 + tn2 * 32 + mloc;
    float v0 = acc0[v], v1 = acc1[v];
    if (EPI == EPI_BIAS_RELU) {
      float bb = bias[m];
      v0 += bb; v1 += bb;
      v0 = v0 > 0.f ? v0 : 0.f; v1 = v1 > 0.f ? v1 : 0.f;
    } else if (EPI == EPI_STARRELU) {
      float r0 = v0 > 0.f ? v0 : 0.f; v0 = ss * r0 * r0 + sb;
      float r1 = v1 > 0.f ? v1 : 0.f; v1 = ss * r1 * r1 + sb;
    }
    Cb[(long)m * ldc + n]      = v0;
    Cb[(long)m * ldc + n + 16] = v1;
  }
}

// ---------------- weight repack for the 3x3 conv: [O][C][9] -> [9][O][C] ----
__global__ __launch_bounds__(256)
void repack_w9(const float* __restrict__ W, float* __restrict__ Wt9, long total, long MK)
{
  long i = (long)blockIdx.x * 256 + threadIdx.x;
  if (i >= total) return;
  int tap = (int)(i % 9);
  long mk = i / 9;
  Wt9[(long)tap * MK + mk] = W[i];
}

// ---------------- 3x3 SAME conv as 9-tap implicit GEMM (WMMA) --------------
// Wt9: [9][O][C] repacked; X: [C][64][64] per batch; out relu(conv+bias).
// BN==64 -> block covers one full image row; dx handled by halo rows in LDS.
__global__ __launch_bounds__(256)
void conv3x3_gemm(const float* __restrict__ Wt9,
                  const float* __restrict__ X, long xBatch,
                  const float* __restrict__ bias,
                  float* __restrict__ C, long cBatch,
                  int M, int K)
{
  __shared__ __align__(16) f16 As[BM * KS];
  __shared__ __align__(16) f16 Bs[66 * KS];   // rows 0 and 65 = zero halo

  const int bz = blockIdx.z;
  const float* Xb = X + (long)bz * xBatch;
  float*       Cb = C + (long)bz * cBatch;

  const int row0 = blockIdx.y * BM;
  const int col0 = blockIdx.x * BN;
  const int y    = col0 >> 6;
  const int tid  = threadIdx.x;
  const int lane = tid & 31;
  const int wv   = tid >> 5;
  const int tm   = wv >> 1, tn2 = wv & 1;
  const int half = lane >> 4, mloc = lane & 15;

  if (tid < 128) {                    // zero halo rows once (k = 0..63)
    int k  = tid & 63;
    int rr = (tid >> 6) ? 65 : 0;
    Bs[rr * KS + k] = (f16)0.f;
  }

  v8f acc0 = {0.f,0.f,0.f,0.f,0.f,0.f,0.f,0.f};
  v8f acc1 = {0.f,0.f,0.f,0.f,0.f,0.f,0.f,0.f};

  const f16* rowA = &As[(tm * 16 + mloc) * KS];

  for (int tap = 0; tap < 9; ++tap) {
    const int dy = tap / 3 - 1, dx = tap % 3 - 1;
    const int yy = y + dy;
    if (yy < 0 || yy > 63) continue;          // whole tap is zero (uniform)
    const float* At = Wt9 + (long)tap * M * K;
    const f16* rowB0 = &Bs[(1 + tn2 * 32 + mloc + dx) * KS];
    const f16* rowB1 = &Bs[(1 + tn2 * 32 + 16 + mloc + dx) * KS];

    for (int k0 = 0; k0 < K; k0 += BK) {
#pragma unroll
      for (int j = 0; j < 4; ++j) {
        int idx = tid + j * 256;
        int r  = idx >> 4;
        int c4 = (idx & 15) * 4;
        const float4 v = *(const float4*)&At[(long)(row0 + r) * K + k0 + c4];
        h4 hv; hv[0] = (f16)v.x; hv[1] = (f16)v.y; hv[2] = (f16)v.z; hv[3] = (f16)v.w;
        *(h4*)&As[r * KS + c4] = hv;
      }
#pragma unroll
      for (int j = 0; j < 4; ++j) {
        int idx = tid + j * 256;
        int kr = idx >> 4;
        int c4 = (idx & 15) * 4;
        const float4 v = *(const float4*)&Xb[(long)(k0 + kr) * NPIX + yy * 64 + c4];
        Bs[(c4 + 1) * KS + kr] = (f16)v.x;    // +1: halo offset
        Bs[(c4 + 2) * KS + kr] = (f16)v.y;
        Bs[(c4 + 3) * KS + kr] = (f16)v.z;
        Bs[(c4 + 4) * KS + kr] = (f16)v.w;
      }
      __syncthreads();

      v16h af0 = load_a_frag(rowA,      half);
      v16h af1 = load_a_frag(rowA + 32, half);
      v16h b00 = load_b_frag(rowB0,      half);
      v16h b01 = load_b_frag(rowB0 + 32, half);
      v16h b10 = load_b_frag(rowB1,      half);
      v16h b11 = load_b_frag(rowB1 + 32, half);
      acc0 = wmma_f16(af0, b00, acc0);
      acc0 = wmma_f16(af1, b01, acc0);
      acc1 = wmma_f16(af0, b10, acc1);
      acc1 = wmma_f16(af1, b11, acc1);
      __syncthreads();
    }
  }

#pragma unroll
  for (int v = 0; v < 8; ++v) {
    int m = row0 + tm * 16 + half * 8 + v;
    int n = col0 + tn2 * 32 + mloc;
    float bb = bias[m];
    float v0 = acc0[v] + bb, v1 = acc1[v] + bb;
    Cb[(long)m * NPIX + n]      = v0 > 0.f ? v0 : 0.f;
    Cb[(long)m * NPIX + n + 16] = v1 > 0.f ? v1 : 0.f;
  }
}

// ---------------- routing head GEMM: out[16,N] = scale*(A @ W^T) + bias ----
// A: [16,K] row-major; W: [N,K] row-major. One 16x16 tile per wave, 8 waves.
__global__ __launch_bounds__(256)
void head_gemm(const float* __restrict__ A, int K,
               const float* __restrict__ Wt,
               const float* __restrict__ bias,
               float scale, float* __restrict__ out, int N)
{
  const int tid  = threadIdx.x;
  const int lane = tid & 31;
  const int wv   = tid >> 5;
  const int n0   = blockIdx.x * 128 + wv * 16;
  const int half = lane >> 4, mloc = lane & 15;

  v8f acc = {0.f,0.f,0.f,0.f,0.f,0.f,0.f,0.f};
  for (int k0 = 0; k0 < K; k0 += 32) {
    const float* ar = A  + (long)mloc * K + k0;
    const float* br = Wt + (long)(n0 + mloc) * K + k0;
    float4 a0 = *(const float4*)(ar + half * 16);
    float4 a1 = *(const float4*)(ar + half * 16 + 4);
    float4 a2 = *(const float4*)(ar + half * 16 + 8);
    float4 a3 = *(const float4*)(ar + half * 16 + 12);
    float4 b0 = *(const float4*)(br + half * 16);
    float4 b1 = *(const float4*)(br + half * 16 + 4);
    float4 b2 = *(const float4*)(br + half * 16 + 8);
    float4 b3 = *(const float4*)(br + half * 16 + 12);
    v16h af, bf;
    af[0]=(f16)a0.x; af[1]=(f16)a0.y; af[2]=(f16)a0.z; af[3]=(f16)a0.w;
    af[4]=(f16)a1.x; af[5]=(f16)a1.y; af[6]=(f16)a1.z; af[7]=(f16)a1.w;
    af[8]=(f16)a2.x; af[9]=(f16)a2.y; af[10]=(f16)a2.z; af[11]=(f16)a2.w;
    af[12]=(f16)a3.x; af[13]=(f16)a3.y; af[14]=(f16)a3.z; af[15]=(f16)a3.w;
    bf[0]=(f16)b0.x; bf[1]=(f16)b0.y; bf[2]=(f16)b0.z; bf[3]=(f16)b0.w;
    bf[4]=(f16)b1.x; bf[5]=(f16)b1.y; bf[6]=(f16)b1.z; bf[7]=(f16)b1.w;
    bf[8]=(f16)b2.x; bf[9]=(f16)b2.y; bf[10]=(f16)b2.z; bf[11]=(f16)b2.w;
    bf[12]=(f16)b3.x; bf[13]=(f16)b3.y; bf[14]=(f16)b3.z; bf[15]=(f16)b3.w;
    acc = wmma_f16(af, bf, acc);
  }
#pragma unroll
  for (int v = 0; v < 8; ++v) {
    int m = half * 8 + v;        // 0..15 == batch index
    int n = n0 + mloc;
    out[(long)m * N + n] = acc[v] * scale + bias[n];
  }
}

// ---------------- spatial mean over 4096 pixels ----------------------------
__global__ __launch_bounds__(256)
void mean_hw(const float* __restrict__ X, float* __restrict__ out)
{
  __shared__ float red[256];
  long base = (long)blockIdx.x * NPIX;
  float s = 0.f;
  for (int i = threadIdx.x; i < NPIX; i += 256) s += X[base + i];
  red[threadIdx.x] = s; __syncthreads();
  for (int o = 128; o > 0; o >>= 1) {
    if (threadIdx.x < o) red[threadIdx.x] += red[threadIdx.x + o];
    __syncthreads();
  }
  if (threadIdx.x == 0) out[blockIdx.x] = red[0] * (1.f / 4096.f);
}

// ---------------- softmax over NF=4 (stride MED) ---------------------------
__global__ void softmax4(const float* __restrict__ logits, float* __restrict__ route)
{
  int i = blockIdx.x * blockDim.x + threadIdx.x;
  if (i >= NB * MEDC) return;
  int b = i >> 10, m = i & 1023;
  const float* L = logits + (long)b * 4096;
  float v0 = L[m], v1 = L[1024 + m], v2 = L[2048 + m], v3 = L[3072 + m];
  float mx = fmaxf(fmaxf(v0, v1), fmaxf(v2, v3));
  float e0 = __expf(v0 - mx), e1 = __expf(v1 - mx), e2 = __expf(v2 - mx), e3 = __expf(v3 - mx);
  float inv = 1.f / (e0 + e1 + e2 + e3);
  float* R = route + (long)b * 4096;
  R[m] = e0 * inv; R[1024 + m] = e1 * inv; R[2048 + m] = e2 * inv; R[3072 + m] = e3 * inv;
}

// ---------------- boundary-corrected tap sums for mean(conv3x3 SAME) -------
__device__ __forceinline__ float blk_reduce(float v, float* red)
{
  int t = threadIdx.x;
  red[t] = v; __syncthreads();
  for (int o = 128; o > 0; o >>= 1) {
    if (t < o) red[t] += red[t + o];
    __syncthreads();
  }
  float r = red[0]; __syncthreads();
  return r;
}

__global__ __launch_bounds__(256)
void tapsums(const float* __restrict__ S, float* __restrict__ T)
{
  __shared__ float red[256];
  long base = (long)blockIdx.x * NPIX;
  float tot = 0.f, r0 = 0.f, r63 = 0.f, c0 = 0.f, c63 = 0.f;
  for (int i = threadIdx.x; i < NPIX; i += 256) {
    float v = S[base + i];
    int y = i >> 6, x = i & 63;
    tot += v;
    if (y == 0)  r0  += v;
    if (y == 63) r63 += v;
    if (x == 0)  c0  += v;
    if (x == 63) c63 += v;
  }
  tot = blk_reduce(tot, red);
  r0  = blk_reduce(r0,  red);
  r63 = blk_reduce(r63, red);
  c0  = blk_reduce(c0,  red);
  c63 = blk_reduce(c63, red);
  if (threadIdx.x == 0) {
    float e00 = S[base], e0N = S[base + 63], eN0 = S[base + 63 * 64], eNN = S[base + 63 * 64 + 63];
    int b = blockIdx.x >> 8, c = blockIdx.x & 255;
    for (int ky = 0; ky < 3; ++ky)
      for (int kx = 0; kx < 3; ++kx) {
        int dy = ky - 1, dx = kx - 1;
        float t = tot;
        if (dy == 1) t -= r0; else if (dy == -1) t -= r63;
        if (dx == 1) t -= c0; else if (dx == -1) t -= c63;
        if (dy != 0 && dx != 0) {
          if (dy == 1)  t += (dx == 1) ? e00 : e0N;
          else          t += (dx == 1) ? eN0 : eNN;
        }
        T[(long)b * 2304 + c * 9 + ky * 3 + kx] = t;
      }
  }
}

// ---------------- fused rfft2 -> dynamic filter -> irfft2 -> gating --------
// One 64x64 plane (b, m) per block, fully resident in LDS (~50 KB).
// In-place: XP holds xp on entry, xp*spw + irfft2(rfft2(xp)*W) on exit.
__global__ __launch_bounds__(256)
void fft_filter_gate(float* __restrict__ XP,
                     const float* __restrict__ route,
                     const float* __restrict__ sroute,
                     const float* __restrict__ cw,    // [64][33][4][2]
                     const float* __restrict__ spw)   // [64][64][4]
{
  __shared__ float plane[NPIX];
  __shared__ float reA[64 * FSZ], imA[64 * FSZ];
  __shared__ float reB[64 * FSZ], imB[64 * FSZ];
  __shared__ float twc[64], tws[64];

  const int pid = blockIdx.x;
  const int b = pid >> 10, m = pid & 1023;
  const long base = (long)pid * NPIX;
  const int tid = threadIdx.x;

  if (tid < 64) {
    float ang = -6.28318530717958647692f * (float)tid / 64.f;
    twc[tid] = __cosf(ang); tws[tid] = __sinf(ang);
  }
  float rc[4], sc[4];
#pragma unroll
  for (int f = 0; f < 4; ++f) {
    rc[f] = route [(long)b * 4096 + f * 1024 + m];
    sc[f] = sroute[(long)b * 4096 + f * 1024 + m];
  }
  for (int i = tid; i < NPIX; i += 256) plane[i] = XP[base + i];
  __syncthreads();

  // rows: real FFT along W
  for (int o = tid; o < 64 * FSZ; o += 256) {
    int h = o / FSZ, kx = o % FSZ;
    const float* pr = &plane[h * 64];
    float sr = 0.f, si = 0.f;
    for (int w = 0; w < 64; ++w) {
      int t = (w * kx) & 63;
      float v = pr[w];
      sr += v * twc[t]; si += v * tws[t];
    }
    reA[o] = sr; imA[o] = si;
  }
  __syncthreads();

  // cols: full FFT along H, then dynamic complex filter (+1/4096 ortho scale)
  for (int o = tid; o < 64 * FSZ; o += 256) {
    int ky = o / FSZ, kx = o % FSZ;
    float sr = 0.f, si = 0.f;
    for (int h = 0; h < 64; ++h) {
      int t = (h * ky) & 63;
      float ar = reA[h * FSZ + kx], ai = imA[h * FSZ + kx];
      sr += ar * twc[t] - ai * tws[t];
      si += ar * tws[t] + ai * twc[t];
    }
    const float* c = cw + (long)(ky * FSZ + kx) * 8;
    float wr = 0.f, wi = 0.f;
#pragma unroll
    for (int f = 0; f < 4; ++f) { wr += rc[f] * c[f * 2]; wi += rc[f] * c[f * 2 + 1]; }
    reB[o] = (sr * wr - si * wi) * (1.f / 4096.f);
    imB[o] = (sr * wi + si * wr) * (1.f / 4096.f);
  }
  __syncthreads();

  // inverse cols: e^{+i t} = twc - i*tws
  for (int o = tid; o < 64 * FSZ; o += 256) {
    int h = o / FSZ, kx = o % FSZ;
    float sr = 0.f, si = 0.f;
    for (int ky = 0; ky < 64; ++ky) {
      int t = (h * ky) & 63;
      float yr = reB[ky * FSZ + kx], yi = imB[ky * FSZ + kx];
      sr += yr * twc[t] + yi * tws[t];
      si += yi * twc[t] - yr * tws[t];
    }
    reA[o] = sr; imA[o] = si;   // reA/imA now hold Z
  }
  __syncthreads();

  // inverse real FFT along W (hermitian) + spatial gating, in place
  for (int i = tid; i < NPIX; i += 256) {
    int h = i >> 6, w = i & 63;
    float s = 0.f;
    for (int kx = 0; kx <= 32; ++kx) {
      int t = (w * kx) & 63;
      float zr = reA[h * FSZ + kx], zi = imA[h * FSZ + kx];
      float re = zr * twc[t] + zi * tws[t];
      s += (kx == 0 || kx == 32) ? re : 2.f * re;
    }
    const float* sp = spw + (long)i * 4;
    float g = 0.f;
#pragma unroll
    for (int f = 0; f < 4; ++f) g += sc[f] * sp[f];
    XP[base + i] = plane[i] * g + s;
  }
}

// ---------------------------------------------------------------------------
extern "C" void kernel_launch(void* const* d_in, const int* in_sizes, int n_in,
                              void* d_out, int out_size, void* d_ws, size_t ws_size,
                              hipStream_t stream)
{
  const float* x         = (const float*)d_in[0];
  const float* w_pw1     = (const float*)d_in[1];
  const float* w_pw2     = (const float*)d_in[2];
  const float* star_sc   = (const float*)d_in[3];
  const float* star_bi   = (const float*)d_in[4];
  const float* rw_fc1_w  = (const float*)d_in[5];
  const float* rw_fc1_b  = (const float*)d_in[6];
  const float* rw_conv_w = (const float*)d_in[7];
  const float* rw_conv_b = (const float*)d_in[8];
  const float* rw_fc2_w  = (const float*)d_in[9];
  const float* rw_fc2_b  = (const float*)d_in[10];
  const float* sp_fc1_w  = (const float*)d_in[11];
  const float* sp_fc1_b  = (const float*)d_in[12];
  const float* sp_fc2_w  = (const float*)d_in[13];
  const float* sp_fc2_b  = (const float*)d_in[14];
  const float* cweights  = (const float*)d_in[15];
  const float* sweights  = (const float*)d_in[16];
  float* out = (float*)d_out;

  // workspace layout
  char* ws = (char*)d_ws;
  float* xp     = (float*)ws;                                   // 256 MB [B][MED][4096]
  float* r1     = (float*)(ws + 268435456);                     //  64 MB [B][256][4096]
  float* r2     = (float*)(ws + 268435456 + 67108864);          //  64 MB
  char*  tail   = ws + 268435456 + 2L * 67108864;
  float* rbar   = (float*)(tail);                               // 16 KB
  float* logits = (float*)(tail + 65536);                       // 256 KB
  float* route  = (float*)(tail + 65536 + 262144);              // 256 KB
  float* sroute = (float*)(tail + 65536 + 2 * 262144);          // 256 KB
  float* Tarr   = (float*)(tail + 65536 + 3 * 262144);          // 144 KB
  float* wt9    = (float*)(tail + 2 * 1024 * 1024);             // 4.5 MB repacked 3x3 weights

  dim3 blk(256);
  long XB = (long)DIMC * NPIX;   // x batch stride
  long HB = (long)HRC  * NPIX;   // 256-channel map batch stride
  long MB = (long)MEDC * NPIX;   // xp batch stride
  long OB = (long)DIMC * NPIX;   // out batch stride

  // --- routing branch (mean hoisted through the final linear layer) ---
  gemm_tiled<EPI_BIAS_RELU><<<dim3(64, HRC / BM, NB), blk, 0, stream>>>(
      rw_fc1_w, 0, DIMC, x, XB, NPIX, rw_fc1_b, nullptr, nullptr,
      r1, HB, NPIX, HRC, NPIX, DIMC);
  gemm_tiled<EPI_BIAS_RELU><<<dim3(64, HRC / BM, NB), blk, 0, stream>>>(
      rw_conv_w, 0, HRC, r1, HB, NPIX, rw_conv_b, nullptr, nullptr,
      r2, HB, NPIX, HRC, NPIX, HRC);
  mean_hw<<<NB * HRC, blk, 0, stream>>>(r2, rbar);
  head_gemm<<<32, blk, 0, stream>>>(rbar, HRC, rw_fc2_w, rw_fc2_b, 1.f, logits, 4096);
  softmax4<<<(NB * MEDC + 255) / 256, blk, 0, stream>>>(logits, route);

  // --- spatial routing branch (3x3 conv + exact boundary-corrected mean) ---
  {
    long MK = (long)HSC * DIMC;           // 256*512
    long total = MK * 9;
    repack_w9<<<(unsigned)((total + 255) / 256), blk, 0, stream>>>(sp_fc1_w, wt9, total, MK);
  }
  conv3x3_gemm<<<dim3(64, HSC / BM, NB), blk, 0, stream>>>(
      wt9, x, XB, sp_fc1_b, r1 /*reuse as s1*/, HB, HSC, DIMC);
  tapsums<<<NB * HSC, blk, 0, stream>>>(r1, Tarr);
  head_gemm<<<32, blk, 0, stream>>>(Tarr, 2304, sp_fc2_w, sp_fc2_b, 1.f / 4096.f, logits, 4096);
  softmax4<<<(NB * MEDC + 255) / 256, blk, 0, stream>>>(logits, sroute);

  // --- main path: pwconv1 + StarReLU (channel-major xp) ---
  gemm_tiled<EPI_STARRELU><<<dim3(64, MEDC / BM, NB), blk, 0, stream>>>(
      w_pw1, 0, DIMC, x, XB, NPIX, nullptr, star_sc, star_bi,
      xp, MB, NPIX, MEDC, NPIX, DIMC);

  // --- fused rfft2 -> dynamic filter -> irfft2 -> spatial gating, in place ---
  fft_filter_gate<<<NB * MEDC, blk, 0, stream>>>(xp, route, sroute, cweights, sweights);

  // --- pwconv2 straight into d_out ([B,DIM,H,W]) ---
  gemm_tiled<EPI_NONE><<<dim3(64, DIMC / BM, NB), blk, 0, stream>>>(
      w_pw2, 0, MEDC, xp, MB, NPIX, nullptr, nullptr, nullptr,
      out, OB, NPIX, DIMC, NPIX, MEDC);
}
